// SGC_85126251807573
// MI455X (gfx1250) — matrix operand order repaired and checked by reading, not compiled, not run on hardware
//
#include <hip/hip_runtime.h>
#include <hip/hip_bf16.h>
#include <cstdint>

typedef __attribute__((ext_vector_type(16))) _Float16 v16h;
typedef __attribute__((ext_vector_type(8)))  _Float16 v8h;
typedef __attribute__((ext_vector_type(4)))  _Float16 v4h;
typedef __attribute__((ext_vector_type(8)))  float    v8f;

#define N_NODES 100000
#define N_EDGES 1600000
#define DIM     128
#define NCLS    40
#define NPAD    48   // 3 x 16 col tiles

__device__ __forceinline__ void atomAddF(float* p, float v) {
  (void)__hip_atomic_fetch_add(p, v, __ATOMIC_RELAXED, __HIP_MEMORY_SCOPE_AGENT);
}

// deg[i] = 1 (self loop)
__global__ void k_deg_init(unsigned* __restrict__ deg) {
  int i = blockIdx.x * 256 + threadIdx.x;
  if (i < N_NODES) deg[i] = 1u;
}

// deg[dst] += 1 per edge
__global__ void k_deg_count(const int* __restrict__ dst, unsigned* __restrict__ deg) {
  int e = blockIdx.x * 256 + threadIdx.x;
  if (e < N_EDGES) atomicAdd(&deg[dst[e]], 1u);
}

// in-place: dinv[i] = rsqrt(deg[i])  (deg >= 1 always)
__global__ void k_dinv(unsigned* __restrict__ deg) {
  int i = blockIdx.x * 256 + threadIdx.x;
  if (i < N_NODES) {
    float d = (float)deg[i];
    ((float*)deg)[i] = rsqrtf(d);
  }
}

// xout[i,:] = dinv[i]^2 * xin[i,:]   (self-loop term; fully initializes xout)
__global__ void k_prop_init(float4* __restrict__ xout, const float4* __restrict__ xin,
                            const float* __restrict__ dinv) {
  int t = blockIdx.x * 256 + threadIdx.x;
  if (t < N_NODES * 32) {
    int node = t >> 5;
    float s = dinv[node];
    s *= s;
    float4 v = xin[t];
    xout[t] = make_float4(s * v.x, s * v.y, s * v.z, s * v.w);
  }
}

// one wave per edge: xout[dst,:] += dinv[src]*dinv[dst] * xin[src,:]
__global__ void __launch_bounds__(256)
k_prop_edges(float* __restrict__ xout, const float* __restrict__ xin,
             const float* __restrict__ dinv,
             const int* __restrict__ src, const int* __restrict__ dstp) {
  int lane = threadIdx.x & 31;
  int e = blockIdx.x * 8 + (threadIdx.x >> 5);
  if (e >= N_EDGES) return;
  int s = src[e];
  int d = dstp[e];
  float nrm = dinv[s] * dinv[d];
  float4 v = ((const float4*)(xin + (size_t)s * DIM))[lane];
  float* o = xout + (size_t)d * DIM + lane * 4;
  atomAddF(o + 0, nrm * v.x);
  atomAddF(o + 1, nrm * v.y);
  atomAddF(o + 2, nrm * v.z);
  atomAddF(o + 3, nrm * v.w);
}

// Fused GEMM (x @ W + b) via v_wmma_f32_16x16x32_f16 + row-wise log_softmax.
// One wave owns a 16-row tile; 3 col-tiles (cols 0..47, zero-padded past 40);
// K=128 as 4 chunks of 32. All operands staged in LDS in WMMA-native layouts
// so operand builds are unconditional ds_load_b128 pairs.
__global__ void __launch_bounds__(256)
k_gemm_lsm(const float* __restrict__ x, const float* __restrict__ Wm,
           const float* __restrict__ bias, float* __restrict__ out) {
  __shared__ alignas(32) _Float16 xs[8][16 * DIM];  // 32 KB: per-wave 16x128 f16 tile
  __shared__ alignas(32) _Float16 wsT[NPAD * DIM];  // 12 KB: W^T f16, [col][K], zero-padded
  __shared__ float bs[NPAD];                        // zero-padded bias

  const int tid  = threadIdx.x;
  const int lane = tid & 31;
  const int wv   = tid >> 5;
  const int tile = blockIdx.x * 8 + wv;
  const int row0 = tile * 16;
  _Float16* myxs = &xs[wv][0];

  // --- stage W^T (f16, [col][K]) and bias, zero-padded to 48 cols -----------
  for (int idx = tid; idx < NPAD * DIM; idx += 256) {
    int col = idx >> 7;        // idx / 128
    int k   = idx & (DIM - 1); // idx % 128
    float w = (col < NCLS) ? Wm[k * NCLS + col] : 0.0f;
    wsT[col * DIM + k] = (_Float16)w;
  }
  if (tid < NPAD) bs[tid] = (tid < NCLS) ? bias[tid] : 0.0f;

  // --- stage this wave's 16x128 tile as f16 (coalesced float4 reads) --------
  for (int r = 0; r < 16; ++r) {
    int gr = row0 + r;
    if (gr > N_NODES - 1) gr = N_NODES - 1;
    float4 v = ((const float4*)(x + (size_t)gr * DIM))[lane];
    v4h hv = {(_Float16)v.x, (_Float16)v.y, (_Float16)v.z, (_Float16)v.w};
    ((v4h*)(myxs + r * DIM))[lane] = hv;
  }
  __syncthreads();

  const int half = (lane >= 16) ? 1 : 0;
  const int rloc = lane & 15;

  v8f acc0 = {}, acc1 = {}, acc2 = {};
#pragma unroll
  for (int kk = 0; kk < 4; ++kk) {
    const int kb = kk * 32;
    // A operand: elems 0..15 -> K = kb + half*16 + 0..15 (contiguous)
    const _Float16* ap = myxs + rloc * DIM + kb + half * 16;
    v8h a_lo = *(const v8h*)(ap);
    v8h a_hi = *(const v8h*)(ap + 8);
    v16h a = __builtin_shufflevector(a_lo, a_hi,
                                     0, 1, 2, 3, 4, 5, 6, 7,
                                     8, 9, 10, 11, 12, 13, 14, 15);
#pragma unroll
    for (int j = 0; j < 3; ++j) {
      const int col = j * 16 + rloc;
      // B operand: elems 0..15 -> K = kb + half*16 + 0..15 (contiguous in wsT)
      v16h bm = *(const v16h*)(wsT + col * DIM + kb + half * 16);
      if (j == 0)
        acc0 = __builtin_amdgcn_wmma_f32_16x16x32_f16(false, a, false, bm, (short)0, acc0, false, false);
      else if (j == 1)
        acc1 = __builtin_amdgcn_wmma_f32_16x16x32_f16(false, a, false, bm, (short)0, acc1, false, false);
      else
        acc2 = __builtin_amdgcn_wmma_f32_16x16x32_f16(false, a, false, bm, (short)0, acc2, false, false);
    }
  }

  // C layout: VGPR i -> row (i + 8*half), lane%16 -> col within tile
  const int col0 = rloc, col1 = rloc + 16, col2 = rloc + 32;
  const float b0 = bs[col0];
  const float b1 = bs[col1];
  const float b2 = bs[col2];
  const bool v2ok = (col2 < NCLS);

#pragma unroll
  for (int i = 0; i < 8; ++i) {
    int row = row0 + i + half * 8;
    float v0 = acc0[i] + b0;
    float v1 = acc1[i] + b1;
    float v2 = acc2[i] + b2;
    // row-wise max over 40 valid cols (16-lane butterfly; halves hold distinct rows)
    float m = fmaxf(v0, v1);
    m = fmaxf(m, v2ok ? v2 : -3.4e38f);
    m = fmaxf(m, __shfl_xor(m, 1, 16));
    m = fmaxf(m, __shfl_xor(m, 2, 16));
    m = fmaxf(m, __shfl_xor(m, 4, 16));
    m = fmaxf(m, __shfl_xor(m, 8, 16));
    float sum = expf(v0 - m) + expf(v1 - m) + (v2ok ? expf(v2 - m) : 0.0f);
    sum += __shfl_xor(sum, 1, 16);
    sum += __shfl_xor(sum, 2, 16);
    sum += __shfl_xor(sum, 4, 16);
    sum += __shfl_xor(sum, 8, 16);
    float lse = m + logf(sum);
    if (row < N_NODES) {
      out[(size_t)row * NCLS + col0] = v0 - lse;
      out[(size_t)row * NCLS + col1] = v1 - lse;
      if (v2ok) out[(size_t)row * NCLS + col2] = v2 - lse;
    }
  }
}

extern "C" void kernel_launch(void* const* d_in, const int* in_sizes, int n_in,
                              void* d_out, int out_size, void* d_ws, size_t ws_size,
                              hipStream_t stream) {
  const float* x  = (const float*)d_in[0];
  const int*   ei = (const int*)d_in[1];   // [2, E] row-major
  const float* Wm = (const float*)d_in[2]; // [128, 40]
  const float* bv = (const float*)d_in[3]; // [40]
  float* out = (float*)d_out;

  const int* src = ei;
  const int* dst = ei + N_EDGES;

  char* ws = (char*)d_ws;
  size_t oA = ((size_t)N_NODES * 4 + 255) & ~(size_t)255;
  size_t oB = oA + (size_t)N_NODES * DIM * 4;
  unsigned* deg = (unsigned*)ws;
  float* dinv = (float*)ws;
  float* xA = (float*)(ws + oA);
  float* xB = (float*)(ws + oB);

  const int gN  = (N_NODES + 255) / 256;
  const int gE  = (N_EDGES + 255) / 256;
  const int gP  = (N_NODES * 32 + 255) / 256;
  const int gEw = (N_EDGES + 7) / 8;

  k_deg_init <<<gN, 256, 0, stream>>>(deg);
  k_deg_count<<<gE, 256, 0, stream>>>(dst, deg);
  k_dinv     <<<gN, 256, 0, stream>>>(deg);

  // hop 1: x -> xA
  k_prop_init <<<gP,  256, 0, stream>>>((float4*)xA, (const float4*)x, dinv);
  k_prop_edges<<<gEw, 256, 0, stream>>>(xA, x, dinv, src, dst);
  // hop 2: xA -> xB
  k_prop_init <<<gP,  256, 0, stream>>>((float4*)xB, (const float4*)xA, dinv);
  k_prop_edges<<<gEw, 256, 0, stream>>>(xB, xA, dinv, src, dst);
  // hop 3: xB -> xA
  k_prop_init <<<gP,  256, 0, stream>>>((float4*)xA, (const float4*)xB, dinv);
  k_prop_edges<<<gEw, 256, 0, stream>>>(xA, xB, dinv, src, dst);

  // classifier + log_softmax
  const int tiles = (N_NODES + 15) / 16;           // 6250
  k_gemm_lsm<<<(tiles + 7) / 8, 256, 0, stream>>>(xA, Wm, bv, out);
}
